// MultiheadAttention_4569845203340
// MI455X (gfx1250) — matrix operand order repaired and checked by reading, not compiled, not run on hardware
//
#include <hip/hip_runtime.h>
#include <hip/hip_bf16.h>
#include <stdint.h>

typedef __attribute__((ext_vector_type(16))) _Float16 v16h;
typedef __attribute__((ext_vector_type(8)))  _Float16 v8h;
typedef __attribute__((ext_vector_type(8)))  float    v8f;

#define DEV __device__ __forceinline__

DEV v8f wmma_f16(v16h a, v16h b, v8f c) {
    return __builtin_amdgcn_wmma_f32_16x16x32_f16(false, a, false, b, (short)0, c, false, false);
}

DEV v16h cat8(v8h lo, v8h hi) {
    return __builtin_shufflevector(lo, hi, 0,1,2,3,4,5,6,7,8,9,10,11,12,13,14,15);
}

// async copy: 4 bytes global -> LDS, per-lane addresses, tracked by ASYNCcnt
DEV void async_g2l_b32(uint32_t lds_byte_addr, const void* gaddr) {
    asm volatile("global_load_async_to_lds_b32 %0, %1, off"
                 :: "v"(lds_byte_addr), "v"(gaddr) : "memory");
}
DEV void wait_async0() { asm volatile("s_wait_asynccnt 0x0" ::: "memory"); }

DEV uint32_t lds_addr(const _Float16* p) {
    return (uint32_t)(size_t)p;   // addrspace(3) offset
}

// ---------------------------------------------------------------------------
// GEMM: [M=8192, K=512] x [512, 512] -> [8192, 512]
// Block = 128 threads (4 waves). Block tile 64x64, wave tile 16x64.
// OUT_MODE: 0 = f32 row-major, 1 = f16 row-major, 2 = f16 transposed per head
//           (Vt layout: [(b*8+h)*64 + hd][s], 4096-wide rows)
// ---------------------------------------------------------------------------
template<bool IN_HALF, int OUT_MODE>
__launch_bounds__(128)
__global__ void gemm_proj(const void* __restrict__ Aptr,
                          const float* __restrict__ W,
                          void* __restrict__ Outp, float scale)
{
    __shared__ __attribute__((aligned(32))) _Float16 lds_a[2048]; // 64M x 32K, A-frag order
    __shared__ __attribute__((aligned(32))) _Float16 lds_w[2048]; // 32K x 64N, B-frag order

    const int tid  = threadIdx.x;
    const int lane = tid & 31;
    const int wv   = tid >> 5;
    const int lm   = lane & 15;
    const int hf   = lane >> 4;
    const int m0   = blockIdx.y * 64;
    const int n0   = blockIdx.x * 64;

    v8f acc[4] = {};

    for (int k0 = 0; k0 < 512; k0 += 32) {
        __syncthreads();
        #pragma unroll
        for (int t = 0; t < 16; ++t) {           // A tile 64x32 -> A-frag layout
            int idx = tid + t * 128;
            int m = idx >> 5, kk = idx & 31;
            _Float16 hv;
            if (IN_HALF) hv = ((const _Float16*)Aptr)[(size_t)(m0 + m) * 512 + k0 + kk];
            else         hv = (_Float16)(((const float*)Aptr)[(size_t)(m0 + m) * 512 + k0 + kk]);
            int strip = m >> 4;
            int L   = (m & 15) + (((kk & 15) >= 8) ? 16 : 0);
            int pos = (kk & 7) + ((kk >= 16) ? 8 : 0);
            lds_a[(strip * 32 + L) * 16 + pos] = hv;
        }
        #pragma unroll
        for (int t = 0; t < 16; ++t) {           // W tile 32x64 -> B-frag layout
            int idx = tid + t * 128;
            int kk = idx >> 6, n = idx & 63;
            _Float16 hw = (_Float16)W[(size_t)(k0 + kk) * 512 + n0 + n];
            int nt  = n >> 4;
            int L   = (n & 15) + (((kk & 15) >= 8) ? 16 : 0);
            int pos = (kk & 7) + ((kk >= 16) ? 8 : 0);
            lds_w[(nt * 32 + L) * 16 + pos] = hw;
        }
        __syncthreads();

        v16h a = *(const v16h*)&lds_a[(wv * 32 + lane) * 16];
        #pragma unroll
        for (int nt = 0; nt < 4; ++nt) {
            v16h bm = *(const v16h*)&lds_w[(nt * 32 + lane) * 16];
            acc[nt] = wmma_f16(a, bm, acc[nt]);
        }
    }

    if (OUT_MODE == 2) {
        // transposed per-head f16 store: rows i are consecutive s -> pack v8h
        #pragma unroll
        for (int nt = 0; nt < 4; ++nt) {
            v8h pk;
            #pragma unroll
            for (int i = 0; i < 8; ++i) pk[i] = (_Float16)(acc[nt][i] * scale);
            int col = n0 + nt * 16 + lm;
            int hh = col >> 6, hd = col & 63;
            int row0 = m0 + wv * 16 + 8 * hf;
            int bb = row0 >> 12, s0 = row0 & 4095;
            *(v8h*)((_Float16*)Outp + ((size_t)((bb * 8 + hh) * 64 + hd) * 4096) + s0) = pk;
        }
    } else {
        #pragma unroll
        for (int nt = 0; nt < 4; ++nt)
            #pragma unroll
            for (int i = 0; i < 8; ++i) {
                int row = m0 + wv * 16 + i + 8 * hf;
                int col = n0 + nt * 16 + lm;
                float v = acc[nt][i] * scale;
                if (OUT_MODE == 1) ((_Float16*)Outp)[(size_t)row * 512 + col] = (_Float16)v;
                else               ((float*)Outp)[(size_t)row * 512 + col] = v;
            }
    }
}

// ---------------------------------------------------------------------------
// Flash attention: grid = (S/64 q-blocks, B*H). Block = 128 threads (4 waves).
// K/V blocks double-buffered in LDS, filled by GLOBAL_LOAD_ASYNC_TO_LDS_B32
// directly in WMMA fragment order (per-lane scatter addresses).
// Bias tile is fed as the WMMA C accumulator of the Q*K^T product.
// ---------------------------------------------------------------------------
__launch_bounds__(128)
__global__ void fa_kernel(const _Float16* __restrict__ Qh,
                          const _Float16* __restrict__ Kh,
                          const _Float16* __restrict__ Vt,   // [(b*8+h)*64+hd][4096]
                          const float* __restrict__ bias,
                          _Float16* __restrict__ Oh)
{
    const int S = 4096, D = 512, HD = 64;
    const int tid = threadIdx.x, lane = tid & 31, wv = tid >> 5;
    const int lm = lane & 15, hf = lane >> 4;
    const int qb = blockIdx.x * 64;
    const int bh = blockIdx.y;
    const int b  = bh >> 3;
    const int h  = bh & 7;

    __shared__ __attribute__((aligned(32))) _Float16 lds_k[2][4096]; // B-frag for Q*K^T
    __shared__ __attribute__((aligned(32))) _Float16 lds_v[2][4096]; // B-frag for P*V
    __shared__ __attribute__((aligned(32))) _Float16 lds_p[4096];    // [wave][16 rows][64 keys]

    // Q A-fragments (rows = qb + 16w + lm)
    const _Float16* Qrow = Qh + ((size_t)(b * S + qb + wv * 16 + lm) * D) + h * HD;
    v16h aQ[2];
    #pragma unroll
    for (int ks = 0; ks < 2; ++ks) {
        const _Float16* p = Qrow + ks * 32 + (hf ? 8 : 0);
        aQ[ks] = cat8(*(const v8h*)p, *(const v8h*)(p + 16));
    }

    v8f o[4] = {};
    float rmax[8], rsum[8];
    #pragma unroll
    for (int i = 0; i < 8; ++i) { rmax[i] = -3e30f; rsum[i] = 0.f; }

    const float* bb = bias + (size_t)(qb + wv * 16 + 8 * hf) * S + lm;

    // async-stage key block jj into buffer jj&1 (fragment-order scatter)
    auto stage = [&](int jj) {
        const int key0 = jj * 64;
        _Float16* bk = lds_k[jj & 1];
        _Float16* bv = lds_v[jj & 1];
        #pragma unroll
        for (int t = 0; t < 16; ++t) {          // K: 2048 dwords, [key][hd] source
            int idx = tid + t * 128;
            int key = idx >> 5, hd = (idx & 31) * 2;
            const void* g = Kh + (size_t)(b * S + key0 + key) * D + h * HD + hd;
            int kt = key >> 4, ks2 = hd >> 5;
            int L  = (key & 15) + (((hd & 15) >= 8) ? 16 : 0);
            async_g2l_b32(lds_addr(&bk[((kt * 2 + ks2) * 32 + L) * 16 + (hd & 7) + (((hd & 31) >= 16) ? 8 : 0)]), g);
        }
        #pragma unroll
        for (int t = 0; t < 16; ++t) {          // V: 2048 dwords, [hd][key] source (Vt)
            int idx = tid + t * 128;
            int hd = idx >> 5, key = (idx & 31) * 2;
            const void* g = Vt + (size_t)(bh * 64 + hd) * 4096 + key0 + key;
            int nt = hd >> 4, ksv = key >> 5;
            int L  = (hd & 15) + (((key & 15) >= 8) ? 16 : 0);
            async_g2l_b32(lds_addr(&bv[((nt * 2 + ksv) * 32 + L) * 16 + (key & 7) + (((key & 31) >= 16) ? 8 : 0)]), g);
        }
    };

    stage(0);
    for (int j = 0; j < 64; ++j) {
        const int key0 = j * 64;
        const int cur = j & 1;

        wait_async0();          // this wave's asyncs for buffer `cur` done
        __syncthreads();        // => everyone's asyncs done; prev buffer free
        if (j + 1 < 64) stage(j + 1);

        // S = Q*K^T + bias  (bias tile rides in as the WMMA C accumulator)
        v8f s[4];
        #pragma unroll
        for (int kt = 0; kt < 4; ++kt) {
            v8f c;
            #pragma unroll
            for (int i = 0; i < 8; ++i)
                c[i] = bb[(size_t)i * S + key0 + kt * 16];
            v16h bK0 = *(const v16h*)&lds_k[cur][((kt * 2 + 0) * 32 + lane) * 16];
            v16h bK1 = *(const v16h*)&lds_k[cur][((kt * 2 + 1) * 32 + lane) * 16];
            c = wmma_f16(aQ[0], bK0, c);
            s[kt] = wmma_f16(aQ[1], bK1, c);
        }

        // online softmax (rows live in 16-lane halves)
        #pragma unroll
        for (int i = 0; i < 8; ++i) {
            float mx = fmaxf(fmaxf(s[0][i], s[1][i]), fmaxf(s[2][i], s[3][i]));
            #pragma unroll
            for (int m = 1; m < 16; m <<= 1) mx = fmaxf(mx, __shfl_xor(mx, m, 32));
            float mn = fmaxf(rmax[i], mx);
            float al = __expf(rmax[i] - mn);
            rmax[i] = mn;
            float ps = 0.f;
            #pragma unroll
            for (int kt = 0; kt < 4; ++kt) {
                float p = __expf(s[kt][i] - mn);
                s[kt][i] = p;
                ps += p;
            }
            #pragma unroll
            for (int m = 1; m < 16; m <<= 1) ps += __shfl_xor(ps, m, 32);
            rsum[i] = rsum[i] * al + ps;
            o[0][i] *= al; o[1][i] *= al; o[2][i] *= al; o[3][i] *= al;
        }

        // P (C layout) -> LDS row-major -> A-fragments (wave-local)
        #pragma unroll
        for (int kt = 0; kt < 4; ++kt)
            #pragma unroll
            for (int i = 0; i < 8; ++i)
                lds_p[(wv * 16 + i + 8 * hf) * 64 + kt * 16 + lm] = (_Float16)s[kt][i];
        asm volatile("s_wait_dscnt 0" ::: "memory");

        v16h aP[2];
        #pragma unroll
        for (int ksv = 0; ksv < 2; ++ksv) {
            const _Float16* pp = &lds_p[(wv * 16 + lm) * 64 + ksv * 32 + (hf ? 8 : 0)];
            aP[ksv] = cat8(*(const v8h*)pp, *(const v8h*)(pp + 16));
        }
        #pragma unroll
        for (int nt = 0; nt < 4; ++nt)
            #pragma unroll
            for (int ksv = 0; ksv < 2; ++ksv) {
                v16h bV = *(const v16h*)&lds_v[cur][((nt * 2 + ksv) * 32 + lane) * 16];
                o[nt] = wmma_f16(aP[ksv], bV, o[nt]);
            }
    }

    #pragma unroll
    for (int nt = 0; nt < 4; ++nt)
        #pragma unroll
        for (int i = 0; i < 8; ++i) {
            int row = qb + wv * 16 + i + 8 * hf;
            Oh[(size_t)(b * S + row) * D + h * HD + nt * 16 + lm] =
                (_Float16)(o[nt][i] * (1.0f / rsum[i]));
        }
}

// ---------------------------------------------------------------------------
extern "C" void kernel_launch(void* const* d_in, const int* in_sizes, int n_in,
                              void* d_out, int out_size, void* d_ws, size_t ws_size,
                              hipStream_t stream) {
    (void)in_sizes; (void)n_in; (void)out_size; (void)ws_size;
    const float* Xq   = (const float*)d_in[0];
    const float* Xm   = (const float*)d_in[1];
    const float* bias = (const float*)d_in[2];
    const float* Wq   = (const float*)d_in[3];
    const float* Wk   = (const float*)d_in[4];
    const float* Wv   = (const float*)d_in[5];
    const float* Wo   = (const float*)d_in[6];

    char* ws = (char*)d_ws;
    const size_t SL = (size_t)2 * 4096 * 512 * sizeof(_Float16); // 8 MB per f16 tensor
    _Float16* Qh = (_Float16*)(ws);
    _Float16* Kh = (_Float16*)(ws + SL);
    _Float16* Vt = (_Float16*)(ws + 2 * SL);   // per-head transposed V
    _Float16* Oh = (_Float16*)(ws + 3 * SL);

    dim3 blk(128);
    dim3 gProj(8, 128);   // N/64 x M/64  (M = B*S = 8192, N = 512)
    const float scale = 0.125f; // 1/sqrt(64) folded into Q

    gemm_proj<false, 1><<<gProj, blk, 0, stream>>>(Xq, Wq, Qh, scale);
    gemm_proj<false, 1><<<gProj, blk, 0, stream>>>(Xm, Wk, Kh, 1.0f);
    gemm_proj<false, 2><<<gProj, blk, 0, stream>>>(Xm, Wv, Vt, 1.0f);

    dim3 gFA(64, 16);     // S/64 q-blocks x (B*H)
    fa_kernel<<<gFA, blk, 0, stream>>>(Qh, Kh, Vt, bias, Oh);

    gemm_proj<true, 0><<<gProj, blk, 0, stream>>>(Oh, Wo, d_out, 1.0f);
}
